// SeqRecModel_24060406792479
// MI455X (gfx1250) — compile-verified, matrix operand order checked by reading, not run on hardware
//
#include <hip/hip_runtime.h>
#include <math.h>

// ---------------------------------------------------------------------------
// Problem constants (from reference)
// ---------------------------------------------------------------------------
#define BB 64
#define SS 32
#define II 64
#define NSESS (BB * SS)          // 2048
#define LL (II - 1)              // 63 GRU steps
#define EE 256
#define HH 128
#define G3H (3 * HH)             // 384

typedef __bf16 bf16_t;
typedef __attribute__((ext_vector_type(16))) __bf16 bf16x16;
typedef __attribute__((ext_vector_type(8)))  __bf16 bf16x8;
typedef __attribute__((ext_vector_type(8)))  float  f32x8;

__device__ __forceinline__ float sigmoidf_(float x) {
    return 1.0f / (1.0f + __expf(-x));
}

// ---------------------------------------------------------------------------
// CDNA5 async global->LDS DMA (ASYNCcnt-tracked), GVS addressing:
//   mem = SADDR(u64) + VADDR(i32) ; LDS dest offset in VDST VGPR.
// ---------------------------------------------------------------------------
__device__ __forceinline__ unsigned lds_off_u32(const void* p) {
    // low 32 bits of a generic pointer into LDS == wave-relative LDS offset
    return (unsigned)(size_t)p;
}

__device__ __forceinline__ void async_load_b128(unsigned lds_off, unsigned goff,
                                                const void* sbase) {
    asm volatile("global_load_async_to_lds_b128 %0, %1, %2"
                 :
                 : "v"(lds_off), "v"(goff), "s"(sbase)
                 : "memory");
}

__device__ __forceinline__ void wait_async0() {
    asm volatile("s_wait_asynccnt 0x0" ::: "memory");
}

// ---------------------------------------------------------------------------
// WMMA helpers (CDNA5 wave32: D[16x16] = A[16x32] x B[32x16] + C, bf16 in, f32 acc)
// A layout (ISA 7.12.2, 16-bit A 16x32): lane&15 = row M; lanes 0-15 hold
// K in {0..7, 16..23}; lanes 16-31 hold K in {8..15, 24..31}.
// For D = X * W^T, the B fragment is W^T (KxN); its storage equals the
// A-style layout applied to W's rows (N as the 16-row dim).
// ---------------------------------------------------------------------------
__device__ __forceinline__ bf16x16 load_frag(const bf16_t* p0, int ld, int lane) {
    int r    = lane & 15;
    int kofs = (lane & 16) ? 8 : 0;
    const bf16_t* p = p0 + r * ld + kofs;
    bf16x8 lo = *(const bf16x8*)(p);
    bf16x8 hi = *(const bf16x8*)(p + 16);
    bf16x16 f;
#pragma unroll
    for (int i = 0; i < 8; ++i) { f[i] = lo[i]; f[i + 8] = hi[i]; }
    return f;
}

// A fragment from an f32-stored LDS tile, converting to bf16 on the fly
// (lowers to ds_load_b128 x4 + v_cvt_pk_bf16_f32).
__device__ __forceinline__ bf16x16 load_frag_f32(const float* p0, int ld, int lane) {
    int r    = lane & 15;
    int kofs = (lane & 16) ? 8 : 0;
    const float* p = p0 + r * ld + kofs;
    bf16x16 f;
#pragma unroll
    for (int i = 0; i < 8; ++i) {
        f[i]     = (bf16_t)p[i];
        f[i + 8] = (bf16_t)p[16 + i];
    }
    return f;
}

__device__ __forceinline__ f32x8 wmma_bf16(bf16x16 a, bf16x16 b, f32x8 c) {
    return __builtin_amdgcn_wmma_f32_16x16x32_bf16(
        /*neg_a=*/false, a, /*neg_b=*/false, b,
        /*c_mod=*/(short)0, c, /*reuse_a=*/false, /*reuse_b=*/false);
}

// ---------------------------------------------------------------------------
// Kernel 1: per-session prep — lengths, target, u, masked input ids
// ---------------------------------------------------------------------------
__global__ void prep_kernel(const int* __restrict__ item_id,
                            const int* __restrict__ eval_from,
                            const int* __restrict__ u_type,
                            int* __restrict__ ids_in,     // [N, LL]
                            int* __restrict__ len_in,     // [N]
                            float* __restrict__ out_target,
                            float* __restrict__ out_u) {
    int n = blockIdx.x * blockDim.x + threadIdx.x;
    if (n >= NSESS) return;
    int b = n / SS, s = n % SS;
    int allow = (s >= eval_from[b]) ? 1 : 0;
    const int* row = item_id + (size_t)n * II;

    int length = 0;
#pragma unroll 4
    for (int i = 0; i < II; ++i) {
        int id = row[i]; id = id < 0 ? 0 : id; id *= allow;
        if (id != 0) ++length;
    }
    int tgt_idx = length - 1; if (tgt_idx < 0) tgt_idx = 0;
    int tgt = row[tgt_idx]; tgt = tgt < 0 ? 0 : tgt; tgt *= allow;
    int lin = length - 1; if (lin < 0) lin = 0;

    len_in[n]     = lin;
    out_target[n] = (float)tgt;
    out_u[n]      = (float)u_type[b];

#pragma unroll 4
    for (int t = 0; t < LL; ++t) {
        int id = row[t]; id = id < 0 ? 0 : id; id *= allow;
        ids_in[(size_t)n * LL + t] = (t < lin) ? id : 0;
    }
}

// ---------------------------------------------------------------------------
// Kernel 2: f32 -> bf16 weight conversion
// ---------------------------------------------------------------------------
__global__ void cvt_bf16_kernel(const float* __restrict__ src,
                                bf16_t* __restrict__ dst, int count) {
    for (int i = blockIdx.x * blockDim.x + threadIdx.x; i < count;
         i += gridDim.x * blockDim.x)
        dst[i] = (bf16_t)src[i];
}

// ---------------------------------------------------------------------------
// Kernel 3: GRU recurrence. 1 block = 16 sessions, 128 threads = 4 waves.
// Wave w owns gate columns [32w, 32w+32) of each gate (6 WMMA D-tiles).
//  - All 72 weight B-fragments are loaded ONCE and kept register-resident
//    across the 63-step loop (~576 VGPRs; wave32 file is 1024).
//  - GRU state h lives in registers in D-fragment coords; LDS keeps only the
//    bf16 copy needed to build A-fragments for h @ W_hh^T.
//  - Embedding gather for step t+1 is an async global->LDS DMA
//    (double-buffered f32 tile) overlapped with the 72 WMMAs of step t.
// ---------------------------------------------------------------------------
__global__ void __launch_bounds__(128)
gru_kernel(const int* __restrict__ ids_in,
           const int* __restrict__ len_in,
           const float* __restrict__ emb,
           const bf16_t* __restrict__ Wih,   // [384,256] bf16 row-major
           const bf16_t* __restrict__ Whh,   // [384,128]
           const float* __restrict__ b_ih,
           const float* __restrict__ b_hh,
           bf16_t* __restrict__ outs,        // [N, LL, HH] bf16
           float* __restrict__ ht) {         // [N, HH] f32
    __shared__ float  xf[2][16 * EE];  // x_t double buffer (f32)  2 x 16 KB
    __shared__ bf16_t hb[16 * HH];     // h tile (bf16)            4 KB
    __shared__ int    slen[16];

    const int n0   = blockIdx.x * 16;
    const int tid  = threadIdx.x;
    const int lane = tid & 31;
    const int wave = tid >> 5;

    for (int i = tid; i < 16 * HH; i += 128) hb[i] = (bf16_t)0.0f;
    if (tid < 16) slen[tid] = len_in[n0 + tid];

    // this thread's gather slice: session m, 32 floats starting at column c0
    const int gm  = tid >> 3;
    const int gc0 = (tid & 7) * 32;

    // issue async gather of x_0 into buffer 0
    {
        int id = ids_in[(size_t)(n0 + gm) * LL + 0];
        unsigned goff = (unsigned)(((size_t)id * EE + gc0) * sizeof(float));
        unsigned loff = lds_off_u32(&xf[0][gm * EE + gc0]);
#pragma unroll
        for (int c8 = 0; c8 < 8; ++c8)
            async_load_b128(loff + c8 * 16, goff + c8 * 16, emb);
    }

    // ---- hoist all weight B-fragments into registers (held across t) ----
    bf16x16 wfi[8][6];   // W_ih: 8 k-tiles x 6 n-tiles
    bf16x16 wfh[4][6];   // W_hh: 4 k-tiles x 6 n-tiles
#pragma unroll
    for (int k = 0; k < 8; ++k)
#pragma unroll
        for (int p = 0; p < 6; ++p) {
            int nrow = (p >> 1) * HH + wave * 32 + (p & 1) * 16;
            wfi[k][p] = load_frag(Wih + (size_t)nrow * EE + k * 32, EE, lane);
        }
#pragma unroll
    for (int k = 0; k < 4; ++k)
#pragma unroll
        for (int p = 0; p < 6; ++p) {
            int nrow = (p >> 1) * HH + wave * 32 + (p & 1) * 16;
            wfh[k][p] = load_frag(Whh + (size_t)nrow * HH + k * 32, HH, lane);
        }

    // biases for this wave's 6 tiles (p = 2*gate + half)
    float bi[6], bh[6];
#pragma unroll
    for (int p = 0; p < 6; ++p) {
        int col = (p >> 1) * HH + wave * 32 + (p & 1) * 16 + (lane & 15);
        bi[p] = b_ih[col];
        bh[p] = b_hh[col];
    }

    // register-resident h in D-fragment coords: h_reg[half][v] for
    // m = v (+8 for hi lanes), j = wave*32 + half*16 + (lane&15)
    float h_reg[2][8];
#pragma unroll
    for (int half = 0; half < 2; ++half)
#pragma unroll
        for (int v = 0; v < 8; ++v) h_reg[half][v] = 0.0f;

    wait_async0();
    __syncthreads();

    for (int t = 0; t < LL; ++t) {
        const float* xcur = xf[t & 1];

        // prefetch x_{t+1} into the other buffer (DMA, overlaps WMMA below)
        if (t + 1 < LL) {
            int id = ids_in[(size_t)(n0 + gm) * LL + (t + 1)];
            unsigned goff = (unsigned)(((size_t)id * EE + gc0) * sizeof(float));
            unsigned loff = lds_off_u32(&xf[(t + 1) & 1][gm * EE + gc0]);
#pragma unroll
            for (int c8 = 0; c8 < 8; ++c8)
                async_load_b128(loff + c8 * 16, goff + c8 * 16, emb);
        }

        f32x8 acc_i[6], acc_h[6];
#pragma unroll
        for (int p = 0; p < 6; ++p)
#pragma unroll
            for (int v = 0; v < 8; ++v) { acc_i[p][v] = bi[p]; acc_h[p][v] = bh[p]; }

        // gi = x_t @ W_ih^T   (K = 256 -> 8 k-tiles, weights in registers)
#pragma unroll
        for (int k = 0; k < 8; ++k) {
            bf16x16 a = load_frag_f32(xcur + k * 32, EE, lane);
#pragma unroll
            for (int p = 0; p < 6; ++p)
                acc_i[p] = wmma_bf16(a, wfi[k][p], acc_i[p]);
        }
        // gh = h @ W_hh^T     (K = 128 -> 4 k-tiles, weights in registers)
#pragma unroll
        for (int k = 0; k < 4; ++k) {
            bf16x16 a = load_frag(hb + k * 32, HH, lane);
#pragma unroll
            for (int p = 0; p < 6; ++p)
                acc_h[p] = wmma_bf16(a, wfh[k][p], acc_h[p]);
        }
        __syncthreads();   // all waves done reading hb before overwrite

        // gate math fully in registers
#pragma unroll
        for (int half = 0; half < 2; ++half) {
            int j = wave * 32 + half * 16 + (lane & 15);
#pragma unroll
            for (int v = 0; v < 8; ++v) {
                int m = (lane < 16) ? v : (8 + v);
                float r  = sigmoidf_(acc_i[0 + half][v] + acc_h[0 + half][v]);
                float z  = sigmoidf_(acc_i[2 + half][v] + acc_h[2 + half][v]);
                float nn = tanhf(acc_i[4 + half][v] + r * acc_h[4 + half][v]);
                float hold  = h_reg[half][v];
                float hnew  = (1.0f - z) * nn + z * hold;
                bool  valid = (t < slen[m]);
                float hnext = valid ? hnew : hold;
                float o     = valid ? hnew : 0.0f;
                h_reg[half][v] = hnext;
                hb[m * HH + j] = (bf16_t)hnext;
                outs[((size_t)(n0 + m) * LL + t) * HH + j] = (bf16_t)o;
            }
        }
        wait_async0();     // x_{t+1} DMA complete
        __syncthreads();   // h + x buffers consistent across waves
    }

    // final hidden state from registers (fragment-coord scatter)
#pragma unroll
    for (int half = 0; half < 2; ++half) {
        int j = wave * 32 + half * 16 + (lane & 15);
#pragma unroll
        for (int v = 0; v < 8; ++v) {
            int m = (lane < 16) ? v : (8 + v);
            ht[(size_t)(n0 + m) * HH + j] = h_reg[half][v];
        }
    }
}

// ---------------------------------------------------------------------------
// Kernel 4: attention readout. 1 block = 16 sessions, 128 threads = 4 waves.
// q2 and the 8 W_a1 fragments kept in registers; per step q1 via WMMA, alpha
// via LDS float atomics, c_loc accumulated in LDS. gru_t tiles are
// double-buffered and prefetched with async global->LDS DMA (bf16).
// ---------------------------------------------------------------------------
__global__ void __launch_bounds__(128)
attn_kernel(const bf16_t* __restrict__ outs,   // [N, LL, HH]
            const float* __restrict__ ht,      // [N, HH]
            const bf16_t* __restrict__ Wa1,    // [128,128] bf16
            const bf16_t* __restrict__ Wa2,    // [128,128] bf16
            const float* __restrict__ W_vt,    // [128]
            const int* __restrict__ len_in,
            float* __restrict__ reps) {        // [N, 2H]
    __shared__ bf16_t gt[2][16 * HH];  // gru_t double buffer  2 x 4 KB
    __shared__ bf16_t htb[16 * HH];    // 4 KB
    __shared__ float  cl[16 * HH];     // 8 KB (c_loc accum)
    __shared__ float  alpha_s[16];
    __shared__ int    slen[16];

    const int n0   = blockIdx.x * 16;
    const int tid  = threadIdx.x;
    const int lane = tid & 31;
    const int wave = tid >> 5;

    for (int i = tid; i < 16 * HH; i += 128) {
        htb[i] = (bf16_t)ht[(size_t)n0 * HH + i];
        cl[i]  = 0.0f;
    }
    if (tid < 16) slen[tid] = len_in[n0 + tid];

    // this thread's gather slice: session m, 16 bf16 starting at column j0
    const int gm  = tid >> 3;
    const int gj0 = (tid & 7) * 16;

    // issue async gather of gru_0 into buffer 0 (2 x b128 per thread)
    {
        unsigned goff = (unsigned)((((size_t)(n0 + gm) * LL + 0) * HH + gj0) * sizeof(bf16_t));
        unsigned loff = lds_off_u32(&gt[0][gm * HH + gj0]);
        async_load_b128(loff, goff, outs);
        async_load_b128(loff + 16, goff + 16, outs);
    }

    // wave w owns column tiles 2w, 2w+1 (j base = wave*32 + p*16)
    float wvt[2];
#pragma unroll
    for (int p = 0; p < 2; ++p)
        wvt[p] = W_vt[wave * 32 + p * 16 + (lane & 15)];

    // hoist W_a1 fragments (4 k-tiles x 2 n-tiles, register-resident)
    bf16x16 wa1f[4][2];
#pragma unroll
    for (int k = 0; k < 4; ++k)
#pragma unroll
        for (int p = 0; p < 2; ++p) {
            int nrow = wave * 32 + p * 16;
            wa1f[k][p] = load_frag(Wa1 + (size_t)nrow * HH + k * 32, HH, lane);
        }

    wait_async0();
    __syncthreads();

    // q2 = ht @ W_a2^T, kept in registers per wave
    f32x8 q2[2];
#pragma unroll
    for (int p = 0; p < 2; ++p)
#pragma unroll
        for (int v = 0; v < 8; ++v) q2[p][v] = 0.0f;
#pragma unroll
    for (int k = 0; k < 4; ++k) {
        bf16x16 a = load_frag(htb + k * 32, HH, lane);
#pragma unroll
        for (int p = 0; p < 2; ++p) {
            int nrow = wave * 32 + p * 16;
            bf16x16 bf = load_frag(Wa2 + (size_t)nrow * HH + k * 32, HH, lane);
            q2[p] = wmma_bf16(a, bf, q2[p]);
        }
    }

    for (int t = 0; t < LL; ++t) {
        const bf16_t* gcur = gt[t & 1];

        if (tid < 16) alpha_s[tid] = 0.0f;
        // prefetch gru_{t+1} into the other buffer
        if (t + 1 < LL) {
            unsigned goff = (unsigned)((((size_t)(n0 + gm) * LL + (t + 1)) * HH + gj0) * sizeof(bf16_t));
            unsigned loff = lds_off_u32(&gt[(t + 1) & 1][gm * HH + gj0]);
            async_load_b128(loff, goff, outs);
            async_load_b128(loff + 16, goff + 16, outs);
        }
        __syncthreads();   // alpha zero visible; gt[t&1] ready from prior drain

        // q1 tiles + sigmoid + dot with W_vt -> alpha partials
        f32x8 acc[2];
#pragma unroll
        for (int p = 0; p < 2; ++p) acc[p] = q2[p];
#pragma unroll
        for (int k = 0; k < 4; ++k) {
            bf16x16 a = load_frag(gcur + k * 32, HH, lane);
#pragma unroll
            for (int p = 0; p < 2; ++p)
                acc[p] = wmma_bf16(a, wa1f[k][p], acc[p]);
        }
#pragma unroll
        for (int v = 0; v < 8; ++v) {
            int m = (lane < 16) ? v : (8 + v);
            float pv = sigmoidf_(acc[0][v]) * wvt[0] + sigmoidf_(acc[1][v]) * wvt[1];
            atomicAdd(&alpha_s[m], pv);
        }
        __syncthreads();

        // c_loc += alpha * mask * gru_t
        {
            float av = (t < slen[gm]) ? alpha_s[gm] : 0.0f;
#pragma unroll
            for (int c = 0; c < 16; ++c)
                cl[gm * HH + gj0 + c] += av * (float)gcur[gm * HH + gj0 + c];
        }
        wait_async0();     // gru_{t+1} DMA complete
        __syncthreads();
    }

    // reps = [c_loc, ht]
    for (int i = tid; i < 16 * HH; i += 128) {
        int m = i >> 7, j = i & 127;
        reps[(size_t)(n0 + m) * (2 * HH) + j]      = cl[i];
        reps[(size_t)(n0 + m) * (2 * HH) + HH + j] = ht[(size_t)n0 * HH + i];
    }
}

// ---------------------------------------------------------------------------
// Host-side launcher
// ---------------------------------------------------------------------------
extern "C" void kernel_launch(void* const* d_in, const int* in_sizes, int n_in,
                              void* d_out, int out_size, void* d_ws, size_t ws_size,
                              hipStream_t stream) {
    const int*   item_id   = (const int*)  d_in[0];
    const int*   eval_from = (const int*)  d_in[1];
    const int*   u_type    = (const int*)  d_in[2];
    const float* emb       = (const float*)d_in[3];
    const float* W_ih      = (const float*)d_in[4];
    const float* W_hh      = (const float*)d_in[5];
    const float* b_ih      = (const float*)d_in[6];
    const float* b_hh      = (const float*)d_in[7];
    const float* W_a1      = (const float*)d_in[8];
    const float* W_a2      = (const float*)d_in[9];
    const float* W_vt      = (const float*)d_in[10];

    char*  ws  = (char*)d_ws;
    size_t off = 0;
    auto alloc = [&](size_t bytes) -> void* {
        void* p = ws + off;
        off = (off + bytes + 255) & ~(size_t)255;
        return p;
    };
    int*    len_in = (int*)   alloc((size_t)NSESS * 4);
    int*    ids_in = (int*)   alloc((size_t)NSESS * LL * 4);
    bf16_t* Wih_b  = (bf16_t*)alloc((size_t)G3H * EE * 2);
    bf16_t* Whh_b  = (bf16_t*)alloc((size_t)G3H * HH * 2);
    bf16_t* Wa1_b  = (bf16_t*)alloc((size_t)HH * HH * 2);
    bf16_t* Wa2_b  = (bf16_t*)alloc((size_t)HH * HH * 2);
    bf16_t* outs   = (bf16_t*)alloc((size_t)NSESS * LL * HH * 2);
    float*  htbuf  = (float*) alloc((size_t)NSESS * HH * 4);

    float* out      = (float*)d_out;
    float* out_reps = out;                                 // [N, 2H]
    float* out_tgt  = out + (size_t)NSESS * 2 * HH;        // [N]
    float* out_u    = out_tgt + NSESS;                     // [N]

    prep_kernel<<<(NSESS + 255) / 256, 256, 0, stream>>>(
        item_id, eval_from, u_type, ids_in, len_in, out_tgt, out_u);

    cvt_bf16_kernel<<<192, 256, 0, stream>>>(W_ih, Wih_b, G3H * EE);
    cvt_bf16_kernel<<<96,  256, 0, stream>>>(W_hh, Whh_b, G3H * HH);
    cvt_bf16_kernel<<<32,  256, 0, stream>>>(W_a1, Wa1_b, HH * HH);
    cvt_bf16_kernel<<<32,  256, 0, stream>>>(W_a2, Wa2_b, HH * HH);

    gru_kernel<<<NSESS / 16, 128, 0, stream>>>(
        ids_in, len_in, emb, Wih_b, Whh_b, b_ih, b_hh, outs, htbuf);

    attn_kernel<<<NSESS / 16, 128, 0, stream>>>(
        outs, htbuf, Wa1_b, Wa2_b, W_vt, len_in, out_reps);

    (void)in_sizes; (void)n_in; (void)out_size; (void)ws_size;
}
